// QSGA_32615981646081
// MI455X (gfx1250) — matrix operand order repaired and checked by reading, not compiled, not run on hardware
//
#include <hip/hip_runtime.h>
#include <hip/hip_bf16.h>

typedef __attribute__((ext_vector_type(16))) _Float16 v16h;
typedef __attribute__((ext_vector_type(8)))  _Float16 v8h;
typedef __attribute__((ext_vector_type(4)))  _Float16 v4h;
typedef __attribute__((ext_vector_type(8)))  float    v8f;

#define NQ   790
#define NS   790
#define NQP  800
#define NSP  800
#define HNUM 8
#define DH   32
#define CDIM 256
#define BQ   2
#define MS   5
#define TILE 512          // halfs per packed 32x16 B tile (32 lanes x 16 halfs)

// ---------------- WMMA fragment loaders ------------------------------------

// A matrix 16x32 f16, row stride ld (halfs). Contiguous K order:
// lanes 0-15: K in {0..15}; lanes 16-31: K in {16..31}.
__device__ __forceinline__ v16h load_A_frag(const _Float16* base, int ld) {
  int lane = threadIdx.x & 31;
  int row  = lane & 15;
  int koff = (lane < 16) ? 0 : 16;
  const _Float16* p = base + row * ld + koff;
  v8h lo = *(const v8h*)(p);
  v8h hi = *(const v8h*)(p + 8);
  v16h a;
#pragma unroll
  for (int j = 0; j < 8; ++j) { a[j] = lo[j]; a[8 + j] = hi[j]; }
  return a;
}

// B matrix 32x16 f16 stored PRE-SWIZZLED in fragment order:
// tile[lane*16 + e] == B[(lane<16?0:16)+e][lane&15].
__device__ __forceinline__ v16h load_B_packed(const _Float16* tile) {
  int lane = threadIdx.x & 31;
  return *(const v16h*)(tile + lane * 16);
}

// ---------------- workspace zeroing ----------------------------------------

__global__ void zero_kernel(unsigned int* p, int n) {
  int i = blockIdx.x * blockDim.x + threadIdx.x;
  if (i < n) p[i] = 0u;
}

// ---------------- projections: X @ W^T + b, f32 in, f16 out ----------------
// mode 0: row-major q layout (scaled by Dh^-0.5)
// mode 1: packed K^T tiles    tile index ((b*H+h)*50 + st), st = s/16
// mode 2: packed V tiles      tile index (((b*H+h)*25 + kt)*2 + nt)
__global__ void proj_kernel(const float* __restrict__ X,
                            const float* __restrict__ W,
                            const float* __restrict__ bias,
                            _Float16* __restrict__ out,
                            int nbatch, int seq, int mode) {
  int tid = blockIdx.x * blockDim.x + threadIdx.x;
  int total = nbatch * seq * CDIM;
  if (tid >= total) return;
  int c = tid & (CDIM - 1);
  int r = tid >> 8;
  int b = r / seq;
  int s = r - b * seq;
  const float4* x4 = (const float4*)(X + (long)r * CDIM);
  const float4* w4 = (const float4*)(W + (long)c * CDIM);
  float acc = bias[c];
#pragma unroll 8
  for (int i = 0; i < CDIM / 4; ++i) {
    float4 a = x4[i], w = w4[i];
    acc += a.x * w.x + a.y * w.y + a.z * w.z + a.w * w.w;
  }
  int h = c >> 5, d = c & 31;
  long idx;
  if (mode == 0) {
    acc *= 0.17677669529663687f;   // fold Dh^-0.5 into q so GEMM1 D needs no VALU
    idx = ((long)(b * HNUM + h) * NQP + s) * DH + d;
  } else if (mode == 1) {
    int st = s >> 4;
    int lane = (s & 15) + (d & 16);
    int e = d & 15;
    idx = ((long)(b * HNUM + h) * 50 + st) * TILE + lane * 16 + e;
  } else {
    int kt = s >> 5, sl = s & 31;
    int nt = d >> 4;
    int lane = (d & 15) + (sl & 16);
    int e = sl & 15;
    idx = (((long)(b * HNUM + h) * 25 + kt) * 2 + nt) * TILE + lane * 16 + e;
  }
  out[idx] = (_Float16)acc;
}

// ---------------- Wo^T to f16, packed B tiles ------------------------------
__global__ void wot_kernel(const float* __restrict__ Wo, _Float16* __restrict__ WoT) {
  int tid = blockIdx.x * blockDim.x + threadIdx.x;   // 65536
  int c = tid >> 8, n = tid & 255;
  int kt = c >> 5, cl = c & 31;
  int ct = n >> 4;
  int lane = (n & 15) + (cl & 16);
  int e = cl & 15;
  WoT[((long)(kt * 16 + ct)) * TILE + lane * 16 + e] = (_Float16)Wo[n * CDIM + c];
}

// ---------------- positional softmax: [H][NQP][NSP] f16 probs --------------
__global__ __launch_bounds__(32)
void pos_kernel(const float* __restrict__ rel,
                const float* __restrict__ Wp,
                const float* __restrict__ bp,
                _Float16* __restrict__ posP) {
  int bid = blockIdx.x;          // h*NQ + q
  int q = bid % NQ, h = bid / NQ;
  int lane = threadIdx.x;
  float w0 = Wp[h * 3 + 0], w1 = Wp[h * 3 + 1], w2 = Wp[h * 3 + 2], bb = bp[h];
  const float* rr = rel + (long)q * NS * 3;
  float loc[25];
  float mx = -3.0e38f;
#pragma unroll
  for (int k = 0; k < 25; ++k) {
    int s = lane + k * 32;
    float l = -3.0e38f;
    if (s < NS) {
      const float* p = rr + s * 3;
      l = w0 * p[0] + w1 * p[1] + w2 * p[2] + bb;
    }
    loc[k] = l;
    mx = fmaxf(mx, l);
  }
#pragma unroll
  for (int off = 16; off >= 1; off >>= 1) mx = fmaxf(mx, __shfl_xor(mx, off, 32));
  float sum = 0.f;
#pragma unroll
  for (int k = 0; k < 25; ++k) {
    float e = __expf(loc[k] - mx);
    loc[k] = e;
    sum += e;
  }
#pragma unroll
  for (int off = 16; off >= 1; off >>= 1) sum += __shfl_xor(sum, off, 32);
  float inv = 1.f / sum;
  _Float16* out = posP + ((long)h * NQP + q) * NSP;
#pragma unroll
  for (int k = 0; k < 25; ++k) {
    int s = lane + k * 32;
    if (s < NS) out[s] = (_Float16)(loc[k] * inv);
  }
  for (int s = NS + lane; s < NSP; s += 32) out[s] = (_Float16)0.f;
}

// ---------------- fused attention: one wave per (b,m,h,q-tile) -------------
__global__ __launch_bounds__(32)
void attn_kernel(const _Float16* __restrict__ qh,
                 const _Float16* __restrict__ kT,
                 const _Float16* __restrict__ vh,
                 const _Float16* __restrict__ posP,
                 const float* __restrict__ gating,
                 _Float16* __restrict__ attO) {
  __shared__ __align__(32) float    S [16 * NSP];   // 51200 B
  __shared__ __align__(32) _Float16 Ah[16 * NSP];   // 25600 B
  int bid = blockIdx.x;
  int qt = bid % 50; bid /= 50;
  int h  = bid % HNUM; bid /= HNUM;
  int m  = bid % MS;   bid /= MS;
  int b  = bid;
  int lane = threadIdx.x;
  int col = lane & 15;
  int rb  = (lane < 16) ? 0 : 8;

  float g  = 1.f / (1.f + __expf(-gating[h]));
  float wp = 1.f - g;

  // ---- GEMM1: scores = (q*scale)_tile @ k^T  (software-pipelined B loads)
  const _Float16* qbase = qh + ((long)(b * HNUM + h) * NQP + qt * 16) * DH;
  v16h a = load_A_frag(qbase, DH);
  const _Float16* kbase = kT + (long)(m * HNUM + h) * 50 * TILE;
  v16h bf = load_B_packed(kbase);
  for (int st = 0; st < 50; ++st) {
    v16h bn = bf;
    if (st + 1 < 50) {
      __builtin_prefetch(kbase + (st + 2) * TILE, 0, 3);
      bn = load_B_packed(kbase + (st + 1) * TILE);
    }
    v8f c = {0.f, 0.f, 0.f, 0.f, 0.f, 0.f, 0.f, 0.f};
    c = __builtin_amdgcn_wmma_f32_16x16x32_f16(false, a, false, bf,
                                               (short)0, c, false, false);
#pragma unroll
    for (int r = 0; r < 8; ++r) S[(rb + r) * NSP + st * 16 + col] = c[r];
    bf = bn;
  }
  __syncthreads();

  // ---- per-row softmax, blend with positional probs, renormalize, to f16
  // vectorized: 4 elements per lane-iteration (b128/b64 LDS traffic)
  for (int row = 0; row < 16; ++row) {
    float4* Sr4 = (float4*)(S + row * NSP);              // 200 float4s
    float mx = -3.0e38f;
    for (int i = lane; i < 200; i += 32) {
      float4 v = Sr4[i];
      int s = i * 4;
      if (s + 0 < NS) mx = fmaxf(mx, v.x);
      if (s + 1 < NS) mx = fmaxf(mx, v.y);
      if (s + 2 < NS) mx = fmaxf(mx, v.z);
      if (s + 3 < NS) mx = fmaxf(mx, v.w);
    }
#pragma unroll
    for (int off = 16; off >= 1; off >>= 1) mx = fmaxf(mx, __shfl_xor(mx, off, 32));
    float sum = 0.f;
    for (int i = lane; i < 200; i += 32) {
      float4 v = Sr4[i];
      int s = i * 4;
      v.x = (s + 0 < NS) ? __expf(v.x - mx) : 0.f;
      v.y = (s + 1 < NS) ? __expf(v.y - mx) : 0.f;
      v.z = (s + 2 < NS) ? __expf(v.z - mx) : 0.f;
      v.w = (s + 3 < NS) ? __expf(v.w - mx) : 0.f;
      Sr4[i] = v;
      sum += v.x + v.y + v.z + v.w;
    }
#pragma unroll
    for (int off = 16; off >= 1; off >>= 1) sum += __shfl_xor(sum, off, 32);
    float inv = wp / sum;                                // fold (1-g) here
    const v4h* pr4 = (const v4h*)(posP + ((long)h * NQP + qt * 16 + row) * NSP);
    float rs = 0.f;
    for (int i = lane; i < 200; i += 32) {
      float4 v = Sr4[i];
      v4h p = pr4[i];                                    // padded pos rows are 0
      v.x = v.x * inv + g * (float)p[0];
      v.y = v.y * inv + g * (float)p[1];
      v.z = v.z * inv + g * (float)p[2];
      v.w = v.w * inv + g * (float)p[3];
      Sr4[i] = v;
      rs += v.x + v.y + v.z + v.w;
    }
#pragma unroll
    for (int off = 16; off >= 1; off >>= 1) rs += __shfl_xor(rs, off, 32);
    float rinv = 1.f / rs;
    v4h* Ar4 = (v4h*)(Ah + row * NSP);
    for (int i = lane; i < 200; i += 32) {
      float4 v = Sr4[i];
      v4h o;
      o[0] = (_Float16)(v.x * rinv);
      o[1] = (_Float16)(v.y * rinv);
      o[2] = (_Float16)(v.z * rinv);
      o[3] = (_Float16)(v.w * rinv);
      Ar4[i] = o;                                        // invalid tail -> 0
    }
  }
  __syncthreads();

  // ---- GEMM2: out = attn @ V (software-pipelined A/B loads)
  v8f acc0 = {0.f, 0.f, 0.f, 0.f, 0.f, 0.f, 0.f, 0.f};
  v8f acc1 = {0.f, 0.f, 0.f, 0.f, 0.f, 0.f, 0.f, 0.f};
  const _Float16* vbase = vh + (long)(m * HNUM + h) * 25 * 2 * TILE;
  v16h af = load_A_frag(Ah, NSP);
  v16h b0 = load_B_packed(vbase);
  v16h b1 = load_B_packed(vbase + TILE);
  for (int kt = 0; kt < 25; ++kt) {
    v16h afn = af, b0n = b0, b1n = b1;
    if (kt + 1 < 25) {
      __builtin_prefetch(vbase + (kt + 2) * 2 * TILE, 0, 3);
      afn = load_A_frag(Ah + (kt + 1) * 32, NSP);
      b0n = load_B_packed(vbase + (kt + 1) * 2 * TILE);
      b1n = load_B_packed(vbase + (kt + 1) * 2 * TILE + TILE);
    }
    acc0 = __builtin_amdgcn_wmma_f32_16x16x32_f16(false, af, false, b0,
                                                  (short)0, acc0, false, false);
    acc1 = __builtin_amdgcn_wmma_f32_16x16x32_f16(false, af, false, b1,
                                                  (short)0, acc1, false, false);
    af = afn; b0 = b0n; b1 = b1n;
  }
  _Float16* obase = attO + ((long)(b * MS + m) * NQP + qt * 16) * CDIM + h * DH;
#pragma unroll
  for (int r = 0; r < 8; ++r) {
    obase[(rb + r) * CDIM + col]      = (_Float16)acc0[r];
    obase[(rb + r) * CDIM + 16 + col] = (_Float16)acc1[r];
  }
}

// ---------------- output projection: attO @ Wo^T + bo ----------------------
__global__ __launch_bounds__(32)
void final_kernel(const _Float16* __restrict__ attO,
                  const _Float16* __restrict__ WoT,
                  const float* __restrict__ bo,
                  float* __restrict__ out) {
  int bid = blockIdx.x;            // 500 row tiles x 16 col tiles
  int ct = bid & 15;
  int rt = bid >> 4;
  int lane = threadIdx.x;
  int col = lane & 15;
  int rb  = (lane < 16) ? 0 : 8;
  const _Float16* Abase = attO + (long)rt * 16 * CDIM;
  v8f acc = {0.f, 0.f, 0.f, 0.f, 0.f, 0.f, 0.f, 0.f};
  v16h af = load_A_frag(Abase, CDIM);
  v16h bf = load_B_packed(WoT + (long)ct * TILE);
  for (int kt = 0; kt < CDIM / 32; ++kt) {
    v16h afn = af, bfn = bf;
    if (kt + 1 < CDIM / 32) {
      afn = load_A_frag(Abase + (kt + 1) * 32, CDIM);
      bfn = load_B_packed(WoT + (long)((kt + 1) * 16 + ct) * TILE);
    }
    acc = __builtin_amdgcn_wmma_f32_16x16x32_f16(false, af, false, bf,
                                                 (short)0, acc, false, false);
    af = afn; bf = bfn;
  }
  float bias = bo[ct * 16 + col];
#pragma unroll
  for (int r = 0; r < 8; ++r) {
    int row = rt * 16 + rb + r;
    int bm = row / NQP;
    int q  = row - bm * NQP;
    if (q < NQ)
      out[((long)bm * NQ + q) * CDIM + ct * 16 + col] = acc[r] + bias;
  }
}

// ---------------------------------------------------------------------------
extern "C" void kernel_launch(void* const* d_in, const int* in_sizes, int n_in,
                              void* d_out, int out_size, void* d_ws, size_t ws_size,
                              hipStream_t stream) {
  const float* q_feat = (const float*)d_in[0];
  const float* s_feat = (const float*)d_in[1];
  const float* rel    = (const float*)d_in[2];
  const float* Wq     = (const float*)d_in[3];
  const float* bq     = (const float*)d_in[4];
  const float* Wk     = (const float*)d_in[5];
  const float* bk     = (const float*)d_in[6];
  const float* Wv     = (const float*)d_in[7];
  const float* bv     = (const float*)d_in[8];
  const float* Wp     = (const float*)d_in[9];
  const float* bp     = (const float*)d_in[10];
  const float* Wo     = (const float*)d_in[11];
  const float* bo     = (const float*)d_in[12];
  const float* gating = (const float*)d_in[13];

  // workspace layout (halfs)
  const size_t n_qh   = (size_t)BQ * HNUM * NQP * DH;         //   409,600
  const size_t n_kT   = (size_t)MS * HNUM * 50 * TILE;        // 1,024,000
  const size_t n_vh   = (size_t)MS * HNUM * 25 * 2 * TILE;    // 1,024,000
  const size_t n_pos  = (size_t)HNUM * NQP * NSP;             // 5,120,000
  const size_t n_attO = (size_t)BQ * MS * NQP * CDIM;         // 2,048,000
  const size_t n_WoT  = (size_t)8 * 16 * TILE;                //    65,536

  _Float16* qh   = (_Float16*)d_ws;
  _Float16* kT   = qh + n_qh;
  _Float16* vh   = kT + n_kT;
  _Float16* posP = vh + n_vh;
  _Float16* attO = posP + n_pos;
  _Float16* WoT  = attO + n_attO;

  size_t totalHalfs = n_qh + n_kT + n_vh + n_pos + n_attO + n_WoT;
  int nwords = (int)(totalHalfs / 2);
  zero_kernel<<<(nwords + 255) / 256, 256, 0, stream>>>((unsigned int*)d_ws, nwords);

  {
    int n = BQ * NQ * CDIM;
    proj_kernel<<<(n + 255) / 256, 256, 0, stream>>>(q_feat, Wq, bq, qh, BQ, NQ, 0);
  }
  {
    int n = MS * NS * CDIM;
    proj_kernel<<<(n + 255) / 256, 256, 0, stream>>>(s_feat, Wk, bk, kT, MS, NS, 1);
    proj_kernel<<<(n + 255) / 256, 256, 0, stream>>>(s_feat, Wv, bv, vh, MS, NS, 2);
  }
  wot_kernel<<<(CDIM * CDIM) / 256, 256, 0, stream>>>(Wo, WoT);
  pos_kernel<<<HNUM * NQ, 32, 0, stream>>>(rel, Wp, bp, posP);
  attn_kernel<<<BQ * MS * HNUM * 50, 32, 0, stream>>>(qh, kT, vh, posP, gating, attO);
  final_kernel<<<(BQ * MS * NQP / 16) * 16, 32, 0, stream>>>(attO, WoT, bo, (float*)d_out);
}
